// WrapperGRU_82360292868656
// MI455X (gfx1250) — compile-verified, matrix operand order checked by reading, not run on hardware
//
#include <hip/hip_runtime.h>
#include <hip/hip_bf16.h>
#include <stdint.h>

#define B_ 64
#define S_ 64
#define T_ 64
#define V_ 32000
#define D_ 512
#define H_ 1024
#define L_ 256
#define H3 (3 * H_)

typedef __attribute__((ext_vector_type(16))) __bf16    v16bf;
typedef __attribute__((ext_vector_type(8)))  float     v8f;
typedef __attribute__((ext_vector_type(4)))  unsigned  u32x4;

#define AS1 __attribute__((address_space(1)))
#define AS3 __attribute__((address_space(3)))

#if __has_builtin(__builtin_amdgcn_global_load_async_to_lds_b32)
#define ASYNC_LDS 1
#else
#define ASYNC_LDS 0
#endif

// ---------------- helpers ----------------
__device__ __forceinline__ unsigned short f2bf(float f) {
  // round-to-nearest-even fp32 -> bf16
  unsigned u = __float_as_uint(f);
  unsigned rnd = ((u >> 16) & 1u) + 0x7fffu;
  return (unsigned short)((u + rnd) >> 16);
}

__device__ __forceinline__ void wait_async_zero() {
#if __has_builtin(__builtin_amdgcn_s_wait_asynccnt)
  __builtin_amdgcn_s_wait_asynccnt(0);
#else
  asm volatile("s_wait_asynccnt 0" ::: "memory");
#endif
}

// ---------------- elementwise kernels ----------------
__global__ void k_f32_to_bf16(const float* __restrict__ src,
                              unsigned short* __restrict__ dst, int n) {
  int i = blockIdx.x * blockDim.x + threadIdx.x;
  if (i < n) dst[i] = f2bf(src[i]);
}

__global__ void k_zero_h(float* __restrict__ hf, unsigned short* __restrict__ hb, int n) {
  int i = blockIdx.x * blockDim.x + threadIdx.x;
  if (i < n) { hf[i] = 0.0f; hb[i] = 0; }
}

// encoder embedding gather: xe[b,s,:] = bf16(E[x[b,s],:])
__global__ void k_enc_gather(const int* __restrict__ x, const float* __restrict__ E,
                             unsigned short* __restrict__ xe, int n) {
  int i = blockIdx.x * blockDim.x + threadIdx.x;
  if (i >= n) return;
  int ti = i / D_;
  int d  = i - ti * D_;
  xe[i] = f2bf(E[(size_t)x[ti] * D_ + d]);
}

// decoder feedback gather: x = (t==0) ? 0 : bf16(E[nxt[b],:])
__global__ void k_dec_gather(const float* __restrict__ E, const int* __restrict__ nxt,
                             unsigned short* __restrict__ xbf, int first) {
  int i = blockIdx.x * blockDim.x + threadIdx.x;
  if (i >= B_ * D_) return;
  int b = i >> 9;          // / D_
  int d = i & (D_ - 1);
  unsigned short v = 0;
  if (!first) v = f2bf(E[(size_t)nxt[b] * D_ + d]);
  xbf[i] = v;
}

// z = mu + eps * exp(0.5*logvar)
__global__ void k_reparam(const float* __restrict__ mu, const float* __restrict__ lv,
                          const float* __restrict__ eps, float* __restrict__ zf,
                          unsigned short* __restrict__ zb, int n) {
  int i = blockIdx.x * blockDim.x + threadIdx.x;
  if (i >= n) return;
  float z = mu[i] + eps[i] * expf(0.5f * lv[i]);
  zf[i] = z;
  zb[i] = f2bf(z);
}

// PyTorch GRUCell gate math (biases already folded into gi/gh by the GEMM)
__global__ void k_gru_cell(const float* __restrict__ gi, const float* __restrict__ gh,
                           float* __restrict__ h, unsigned short* __restrict__ hb, int n) {
  int i = blockIdx.x * blockDim.x + threadIdx.x;
  if (i >= n) return;
  int b = i / H_;
  int j = i - b * H_;
  const float* gib = gi + (size_t)b * H3;
  const float* ghb = gh + (size_t)b * H3;
  float ir = gib[j], iz = gib[H_ + j], in = gib[2 * H_ + j];
  float hr = ghb[j], hz = ghb[H_ + j], hn = ghb[2 * H_ + j];
  float r = 1.0f / (1.0f + expf(-(ir + hr)));
  float z = 1.0f / (1.0f + expf(-(iz + hz)));
  float nn = tanhf(in + r * hn);
  float hv = (1.0f - z) * nn + z * h[i];
  h[i] = hv;
  hb[i] = f2bf(hv);
}

// per-row argmax over V columns, first-max tie-break (jnp.argmax semantics)
__global__ void k_argmax(const float* __restrict__ logits, long ldc,
                         int* __restrict__ nxt, int Vn) {
  __shared__ float sv[256];
  __shared__ int   si[256];
  int b = blockIdx.x;
  const float* row = logits + (size_t)b * ldc;
  int tid = threadIdx.x;
  float best = -3.402823466e38f;
  int bi = 0;
  for (int v = tid; v < Vn; v += blockDim.x) {
    float xv = row[v];
    if (xv > best) { best = xv; bi = v; }
  }
  sv[tid] = best; si[tid] = bi;
  __syncthreads();
  for (int s = 128; s > 0; s >>= 1) {
    if (tid < s) {
      float ov = sv[tid + s]; int oi = si[tid + s];
      if (ov > sv[tid] || (ov == sv[tid] && oi < si[tid])) { sv[tid] = ov; si[tid] = oi; }
    }
    __syncthreads();
  }
  if (tid == 0) nxt[b] = si[0];
}

// ---------------- WMMA bf16 GEMM: C[M,N] = A[M,K] * W[N,K]^T + bias ----------------
// Block = 256 threads = 8 waves; block tile = 16 (M) x 512 (N); wave tile = 16x64
// (4 accumulators sharing one A fragment). The 16xK A panel is staged once per
// block into LDS (async global->LDS when available, ASYNCcnt-tracked), then each
// K=32 slab costs: 2 ds_load_b128 (A) + 8 global b128 (4 W frags) + 4 WMMAs.
// Operand layouts per CDNA5 ISA 7.12.2 (wave32):
//   lane = (row 0..15) | (khalf<<4); v16bf elem e: K = khalf*8 + e (e<8), else 16+khalf*8+(e-8)
__global__ void k_gemm_bf16(const unsigned short* __restrict__ A, int lda,
                            const unsigned short* __restrict__ W,
                            const float* __restrict__ bias,
                            float* __restrict__ C, long ldc,
                            int M, int N, int K, int nb) {
  __shared__ unsigned short sA[16 * 1024];   // 32 KB: 16 rows x K<=1024 bf16

  int mtile = blockIdx.x / nb;
  int nblk  = blockIdx.x - mtile * nb;
  int m0    = mtile << 4;
  int tid   = threadIdx.x;

  // ---- stage A panel [16][K] into LDS (whole block) ----
  {
    int dwK   = K >> 1;          // dwords per row
    int total = 16 * dwK;        // always a multiple of 256 here (K >= 32)
    for (int q = tid; q < total; q += 256) {
      int row = q / dwK;
      int col = q - row * dwK;   // dword index within row
      const unsigned short* gsrc = A + (size_t)(m0 + row) * lda + 2 * col;
      unsigned short*       ldst = &sA[row * K + 2 * col];
#if ASYNC_LDS
      __builtin_amdgcn_global_load_async_to_lds_b32(
          (AS1 int*)(unsigned long long)gsrc,
          (AS3 int*)(unsigned)(unsigned long long)ldst, 0, 0);
#else
      *(unsigned*)ldst = *(const unsigned*)gsrc;
#endif
    }
#if ASYNC_LDS
    wait_async_zero();
#endif
  }
  __syncthreads();

  // ---- per-wave 16x64 tile ----
  int wv = tid >> 5;
  int n0 = (nblk * 8 + wv) << 6;            // wave's N base (64-wide)
  if (n0 >= N) return;                      // wave-uniform exit (after the only barrier)

  int lane = tid & 31;
  int rr   = lane & 15;                     // M row (A) / N col (B)
  int kh   = (lane >> 4) << 3;              // 0 or 8: K-half select
  int rowbase = rr * K;

  const unsigned short* Wr[4];
#pragma unroll
  for (int f = 0; f < 4; ++f) Wr[f] = W + (size_t)(n0 + 16 * f + rr) * K;

  v8f acc[4] = {{}, {}, {}, {}};
  union Frag { v16bf v; u32x4 q[2]; };

  for (int kb = 0; kb < K; kb += 32) {
    int base = kb + kh;
    if (kb + 32 < K) {
#pragma unroll
      for (int f = 0; f < 4; ++f)
        __builtin_prefetch(Wr[f] + base + 32, 0, 1);   // global_prefetch_b8
    }
    Frag a;
    a.q[0] = *(const u32x4*)&sA[rowbase + base];        // ds_load_b128
    a.q[1] = *(const u32x4*)&sA[rowbase + base + 16];
    Frag w[4];
#pragma unroll
    for (int f = 0; f < 4; ++f) {
      w[f].q[0] = *(const u32x4*)(Wr[f] + base);        // global_load_b128
      w[f].q[1] = *(const u32x4*)(Wr[f] + base + 16);
    }
#pragma unroll
    for (int f = 0; f < 4; ++f)
      acc[f] = __builtin_amdgcn_wmma_f32_16x16x32_bf16(false, a.v, false, w[f].v,
                                                       (short)0, acc[f], false, false);
  }

  int mtop = (lane >> 4) << 3;              // C/D layout: lanes 16..31 hold M=8..15
#pragma unroll
  for (int f = 0; f < 4; ++f) {
    int nc = n0 + 16 * f + rr;
    float bv = bias ? bias[nc] : 0.0f;
#pragma unroll
    for (int vq = 0; vq < 8; ++vq)
      C[(size_t)(m0 + mtop + vq) * ldc + nc] = acc[f][vq] + bv;
  }
}

// ---------------- host-side orchestration ----------------
extern "C" void kernel_launch(void* const* d_in, const int* in_sizes, int n_in,
                              void* d_out, int out_size, void* d_ws, size_t ws_size,
                              hipStream_t stream) {
  const int*   x     = (const int*)  d_in[0];
  const float* eps   = (const float*)d_in[1];
  const float* E     = (const float*)d_in[2];
  const float* Wihe  = (const float*)d_in[3];
  const float* Whhe  = (const float*)d_in[4];
  const float* bihe  = (const float*)d_in[5];
  const float* bhhe  = (const float*)d_in[6];
  const float* Wmu   = (const float*)d_in[7];
  const float* bmu   = (const float*)d_in[8];
  const float* Wlv   = (const float*)d_in[9];
  const float* blv   = (const float*)d_in[10];
  const float* Wproj = (const float*)d_in[11];
  const float* bproj = (const float*)d_in[12];
  const float* Wihd  = (const float*)d_in[13];
  const float* Whhd  = (const float*)d_in[14];
  const float* bihd  = (const float*)d_in[15];
  const float* bhhd  = (const float*)d_in[16];
  const float* Wout  = (const float*)d_in[17];
  const float* bout  = (const float*)d_in[18];
  float* out = (float*)d_out;

  // ---- workspace carve-up ----
  char* ws = (char*)d_ws;
  size_t off = 0;
  auto carve = [&](size_t bytes) -> void* {
    void* p = ws + off;
    off = (off + bytes + 255) & ~(size_t)255;
    return p;
  };
  unsigned short* wihe_b  = (unsigned short*)carve((size_t)H3 * D_ * 2);
  unsigned short* whhe_b  = (unsigned short*)carve((size_t)H3 * H_ * 2);
  unsigned short* wihd_b  = (unsigned short*)carve((size_t)H3 * D_ * 2);
  unsigned short* whhd_b  = (unsigned short*)carve((size_t)H3 * H_ * 2);
  unsigned short* wout_b  = (unsigned short*)carve((size_t)V_ * H_ * 2);
  unsigned short* wmu_b   = (unsigned short*)carve((size_t)L_ * H_ * 2);
  unsigned short* wlv_b   = (unsigned short*)carve((size_t)L_ * H_ * 2);
  unsigned short* wproj_b = (unsigned short*)carve((size_t)H_ * L_ * 2);
  unsigned short* xe_b    = (unsigned short*)carve((size_t)B_ * S_ * D_ * 2);
  float*          h_f     = (float*)         carve((size_t)B_ * H_ * 4);
  unsigned short* h_b     = (unsigned short*)carve((size_t)B_ * H_ * 2);
  float*          gi_f    = (float*)         carve((size_t)B_ * H3 * 4);
  float*          gh_f    = (float*)         carve((size_t)B_ * H3 * 4);
  float*          z_f     = (float*)         carve((size_t)B_ * L_ * 4);
  unsigned short* z_b     = (unsigned short*)carve((size_t)B_ * L_ * 2);
  unsigned short* x_b     = (unsigned short*)carve((size_t)B_ * D_ * 2);
  int*            nxt     = (int*)           carve((size_t)B_ * 4);
  if (off > ws_size) return;  // workspace too small: bail (deterministic)

  auto conv = [&](const float* s, unsigned short* d, int n) {
    k_f32_to_bf16<<<(n + 255) / 256, 256, 0, stream>>>(s, d, n);
  };
  auto gemm = [&](const unsigned short* A, int lda, const unsigned short* Wb,
                  const float* bias, float* C, long ldc, int M, int N, int K) {
    int nb     = (N / 64 + 7) / 8;       // N is always a multiple of 64 here
    int blocks = (M / 16) * nb;
    k_gemm_bf16<<<blocks, 256, 0, stream>>>(A, lda, Wb, bias, C, ldc, M, N, K, nb);
  };

  // ---- one-time conversions (per launch; deterministic) ----
  conv(Wihe,  wihe_b,  H3 * D_);
  conv(Whhe,  whhe_b,  H3 * H_);
  conv(Wihd,  wihd_b,  H3 * D_);
  conv(Whhd,  whhd_b,  H3 * H_);
  conv(Wout,  wout_b,  V_ * H_);   // 65 MB bf16: resident in 192 MB L2 across decoder steps
  conv(Wmu,   wmu_b,   L_ * H_);
  conv(Wlv,   wlv_b,   L_ * H_);
  conv(Wproj, wproj_b, H_ * L_);

  // ---- encoder ----
  {
    int n = B_ * S_ * D_;
    k_enc_gather<<<(n + 255) / 256, 256, 0, stream>>>(x, E, xe_b, n);
  }
  k_zero_h<<<(B_ * H_ + 255) / 256, 256, 0, stream>>>(h_f, h_b, B_ * H_);
  for (int t = 0; t < S_; ++t) {
    gemm(xe_b + (size_t)t * D_, S_ * D_, wihe_b, bihe, gi_f, H3, B_, H3, D_);
    gemm(h_b, H_, whhe_b, bhhe, gh_f, H3, B_, H3, H_);
    k_gru_cell<<<(B_ * H_ + 255) / 256, 256, 0, stream>>>(gi_f, gh_f, h_f, h_b, B_ * H_);
  }

  // ---- latent heads -> d_out tail ----
  float* mu_out = out + (size_t)B_ * T_ * V_;
  float* lv_out = mu_out + (size_t)B_ * L_;
  gemm(h_b, H_, wmu_b, bmu, mu_out, L_, B_, L_, H_);
  gemm(h_b, H_, wlv_b, blv, lv_out, L_, B_, L_, H_);
  k_reparam<<<(B_ * L_ + 255) / 256, 256, 0, stream>>>(mu_out, lv_out, eps, z_f, z_b, B_ * L_);

  // ---- decoder init: h0 = z @ Wproj^T + bproj ----
  gemm(z_b, L_, wproj_b, bproj, h_f, H_, B_, H_, L_);
  conv(h_f, h_b, B_ * H_);

  // ---- autoregressive decode ----
  const long ldl = (long)T_ * V_;  // logits row stride inside [B,T,V]
  for (int t = 0; t < T_; ++t) {
    k_dec_gather<<<(B_ * D_ + 255) / 256, 256, 0, stream>>>(E, nxt, x_b, t == 0 ? 1 : 0);
    gemm(x_b, D_, wihd_b, bihd, gi_f, H3, B_, H3, D_);
    gemm(h_b, H_, whhd_b, bhhd, gh_f, H3, B_, H3, H_);
    k_gru_cell<<<(B_ * H_ + 255) / 256, 256, 0, stream>>>(gi_f, gh_f, h_f, h_b, B_ * H_);
    float* logits_t = out + (size_t)t * V_;
    gemm(h_b, H_, wout_b, bout, logits_t, ldl, B_, V_, H_);
    k_argmax<<<B_, 256, 0, stream>>>(logits_t, ldl, nxt, V_);
  }
}
